// LocalSpatialEncodingModule_73160472920435
// MI455X (gfx1250) — matrix operand order repaired and checked by reading, not compiled, not run on hardware
//
#include <hip/hip_runtime.h>

namespace {

constexpr int   kB   = 8;
constexpr int   kN   = 4096;
constexpr int   kNS  = 32;    // == wave32: lane <-> sample
constexpr int   kC1  = 32;
constexpr int   kC2  = 64;
constexpr int   kCO  = kC2 + 1;          // 65 output channels
constexpr int   kPPW = 8;                // points per wave (amortize weights)
constexpr float kR2  = 0.25f * 0.25f;

typedef __attribute__((ext_vector_type(16))) _Float16 v16h;
typedef __attribute__((ext_vector_type(8)))  float    v8f;

// ---------------------------------------------------------------------------
// Kernel 1: ball query. One block of 256 threads handles 256 query points of
// one batch; the whole batch's xyz (48 KB) lives in LDS (WGP has 320 KB).
// Ascending index scan with early exit at 32 hits == reference top_k semantics
// (top_k over negated index-keys == first 32 valid indices in ascending order).
// Also writes the density channel (= #non-pad samples / 64).
// ---------------------------------------------------------------------------
__global__ __launch_bounds__(256) void ball_query_kernel(
    const float* __restrict__ xyz, int* __restrict__ idxbuf,
    float* __restrict__ out) {
  __shared__ float sx[kN], sy[kN], sz[kN];
  const int b = blockIdx.y;
  const float* base = xyz + (size_t)b * kN * 3;
  for (int i = threadIdx.x; i < kN; i += 256) {
    sx[i] = base[3 * i + 0];
    sy[i] = base[3 * i + 1];
    sz[i] = base[3 * i + 2];
  }
  __syncthreads();

  const int n = blockIdx.x * 256 + threadIdx.x;
  const float qx = sx[n], qy = sy[n], qz = sz[n];
  int* myidx = idxbuf + ((size_t)b * kN + n) * kNS;

  int cnt = 0;
  int firstIdx = n;  // self is always within radius (d2 == 0), so cnt >= 1
  for (int m = 0; m < kN && cnt < kNS; ++m) {
    const float dx = sx[m] - qx;
    const float dy = sy[m] - qy;
    const float dz = sz[m] - qz;
    const float d2 = dx * dx + dy * dy + dz * dz;
    if (d2 < kR2) {
      if (cnt == 0) firstIdx = m;
      myidx[cnt++] = m;
    }
  }
  for (int s = cnt; s < kNS; ++s) myidx[s] = firstIdx;  // pad like reference

  // unique_cnt == number of non-pad entries == cnt (stored indices distinct)
  out[((size_t)b * kN + n) * kCO + kC2] = (float)cnt / (float)kC2;
}

// ---------------------------------------------------------------------------
// Kernel 2: gather + 2-layer MLP + max-pool. One wave handles kPPW points;
// loop-invariant weight fragments are hoisted into registers.
// Layer 1 (3 -> 32) is computed per-lane *directly into the WMMA A-fragment
// layout* (16-bit A 16x32: lane<16 holds K {0..7,16..23}, lane>=16 holds
// K {8..15,24..31}, row M = lane%16). Layer 2 (32x32 @ 32x64) is 8
// v_wmma_f32_16x16x32_f16 ops (2 M-tiles x 4 N-tiles, K=32 in one shot).
// ---------------------------------------------------------------------------
__global__ __launch_bounds__(256) void mlp_pool_kernel(
    const float* __restrict__ xyz, const float* __restrict__ w1,
    const float* __restrict__ b1, const float* __restrict__ w2,
    const float* __restrict__ b2, const int* __restrict__ idxbuf,
    float* __restrict__ out) {
  const int wave = (blockIdx.x * 256 + threadIdx.x) >> 5;
  const int lane = threadIdx.x & 31;
  const int  nb = lane & 15;
  const bool hi = lane >= 16;

  // --- Hoisted B fragments: w2^T as 4 tiles of K=32 x N=16 (16-bit B layout:
  // lane<16 -> column n=lane, K {0..7,16..23}; lane>=16 -> column n=lane-16, K {8..15,24..31})
  v16h bf[4];
  {
    const int kb = hi ? 8 : 0;
#pragma unroll
    for (int t = 0; t < 4; ++t) {
      const float* wrow = w2 + (size_t)(t * 16 + nb) * kC1 + kb;  // w2[n][k..]
#pragma unroll
      for (int j = 0; j < 16; ++j) bf[t][j] = (_Float16)wrow[j < 8 ? j : j + 8];
    }
  }

  // --- Hoisted layer-1 weights for exactly the 16 K-slots this lane owns
  const int kbase = hi ? 8 : 0;
  float w1x[16], w1y[16], w1z[16], b1v[16];
#pragma unroll
  for (int j = 0; j < 16; ++j) {
    const int k = kbase + (j < 8 ? j : j + 8);
    w1x[j] = w1[3 * k + 0];
    w1y[j] = w1[3 * k + 1];
    w1z[j] = w1[3 * k + 2];
    b1v[j] = b1[k];
  }
  // --- Hoisted b2 values this lane needs (column n = t*16 + nb)
  float b2v[4];
#pragma unroll
  for (int t = 0; t < 4; ++t) b2v[t] = b2[t * 16 + nb];

  for (int pp = 0; pp < kPPW; ++pp) {
    const int p = wave * kPPW + pp;       // global point id
    const int b = p >> 12;                // p / kN
    const int n = p & (kN - 1);           // p % kN

    const float* base = xyz + (size_t)b * kN * 3;
    const int gi = idxbuf[(size_t)p * kNS + lane];

    const float cx = base[3 * n + 0];
    const float cy = base[3 * n + 1];
    const float cz = base[3 * n + 2];
    const float rx = base[3 * gi + 0] - cx;   // rel coords, sample s == lane
    const float ry = base[3 * gi + 1] - cy;
    const float rz = base[3 * gi + 2] - cz;

    // --- A fragments: layer-1 activations built in fragment layout
    v16h af[2];
#pragma unroll
    for (int mt = 0; mt < 2; ++mt) {
      const int srow = mt * 16 + nb;       // sample row this lane represents
      const float ax = __shfl(rx, srow, 32);
      const float ay = __shfl(ry, srow, 32);
      const float az = __shfl(rz, srow, 32);
#pragma unroll
      for (int j = 0; j < 16; ++j) {
        const float h = b1v[j] + ax * w1x[j] + ay * w1y[j] + az * w1z[j];
        af[mt][j] = (_Float16)fmaxf(h, 0.0f);   // relu
      }
    }

    // --- Layer 2: 8 WMMAs, f32 accumulate
    const v8f zero = {};
    v8f acc[2][4];
#pragma unroll
    for (int mt = 0; mt < 2; ++mt)
#pragma unroll
      for (int t = 0; t < 4; ++t)
        acc[mt][t] = __builtin_amdgcn_wmma_f32_16x16x32_f16(
            false, af[mt], false, bf[t], (short)0, zero, false, false);

    // --- Max-pool over samples: rows M live in the 8 accumulator VGPRs, the
    // two M-tiles and the two lane halves. Bias+relu commute with the max
    // (b2 constant per column, relu monotone).
#pragma unroll
    for (int t = 0; t < 4; ++t) {
      float vmax = fmaxf(acc[0][t][0], acc[1][t][0]);
#pragma unroll
      for (int j = 1; j < 8; ++j)
        vmax = fmaxf(vmax, fmaxf(acc[0][t][j], acc[1][t][j]));
      vmax = fmaxf(vmax + b2v[t], 0.0f);
      vmax = fmaxf(vmax, __shfl_xor(vmax, 16, 32));  // merge M 0-7 vs 8-15
      if (!hi) out[(size_t)p * kCO + t * 16 + nb] = vmax;
    }
  }
}

}  // namespace

extern "C" void kernel_launch(void* const* d_in, const int* in_sizes, int n_in,
                              void* d_out, int out_size, void* d_ws,
                              size_t ws_size, hipStream_t stream) {
  (void)in_sizes; (void)n_in; (void)out_size; (void)ws_size;
  const float* xyz = (const float*)d_in[0];  // B x N x 3
  const float* w1  = (const float*)d_in[1];  // 32 x 3
  const float* b1  = (const float*)d_in[2];  // 32
  const float* w2  = (const float*)d_in[3];  // 64 x 32
  const float* b2  = (const float*)d_in[4];  // 64
  float* out  = (float*)d_out;               // B x N x 65
  int* idxbuf = (int*)d_ws;                  // B*N*32 ints = 4 MB scratch

  dim3 g1(kN / 256, kB);
  ball_query_kernel<<<g1, 256, 0, stream>>>(xyz, idxbuf, out);

  const int nPoints = kB * kN;               // 32768 points
  const int nWaves  = nPoints / kPPW;        // 4096 waves
  const int nBlocks = nWaves / 8;            // 8 waves per 256-thread block
  mlp_pool_kernel<<<nBlocks, 256, 0, stream>>>(xyz, w1, b1, w2, b2, idxbuf,
                                               out);
}
